// New4_77395310674433
// MI455X (gfx1250) — compile-verified, matrix operand order checked by reading, not run on hardware
//
#include <hip/hip_runtime.h>
#include <hip/hip_bf16.h>

typedef __attribute__((ext_vector_type(16))) _Float16 v16h;
typedef __attribute__((ext_vector_type(8)))  _Float16 v8h;
typedef __attribute__((ext_vector_type(8)))  float    v8f;

#define ITEM_NUM 50000
#define EMBED    256
#define HIDDEN   128
#define BATCH    4096
#define HIST     50
#define K_NEAR   10
#define D4       64
#define D2       128

// -------------------------------------------------------------------------
// Kernel 0: repack W1 (HIDDEN x EMBED, row-major f32) into fp16 WMMA
// B-fragment layout for v_wmma_f32_16x16x32_f16 (wave32).
// B tile is 32x16 (K x N) at (k0 = ks*32, n0 = ntile*16).
// lane L: column c = L&15, group g = L>>4.
// half i: K = (i<8 ? 0 : 16) + g*8 + (i&7).
// frag flat: [ntile(8)][ks(8)][lane(32)][i(16)] halves.
// -------------------------------------------------------------------------
__global__ __launch_bounds__(256) void w1_to_frag(const float* __restrict__ W1,
                                                  _Float16* __restrict__ frag) {
    int fi    = blockIdx.x * blockDim.x + threadIdx.x;   // 0..32767
    int i     = fi & 15;
    int lane  = (fi >> 4) & 31;
    int ks    = (fi >> 9) & 7;
    int ntile = fi >> 12;
    int g = lane >> 4;
    int c = lane & 15;
    int n = ntile * 16 + c;
    int k = ks * 32 + ((i < 8) ? 0 : 16) + g * 8 + (i & 7);
    frag[fi] = (_Float16)W1[n * EMBED + k];              // B[k][n] = W1[n][k]
}

// -------------------------------------------------------------------------
// Kernel 1: region attention. One wave (32 lanes) per POI, 8 POIs per block.
// Reproduces the reference's flat-reinterpret "k = kv.reshape(N, 64, 10)":
// score_j = (1/8) * sum_a q[a] * kvflat[a*10 + j]
// All 10 butterfly-reduce chains are independent -> pipelined bpermutes.
// -------------------------------------------------------------------------
__device__ __forceinline__ void attn_one(const float* __restrict__ qflat,
                                         const float* __restrict__ kvflat,
                                         float* __restrict__ out, int lane) {
    float qa0 = qflat[lane];
    float qa1 = qflat[lane + 32];
    float s[K_NEAR];
#pragma unroll
    for (int j = 0; j < K_NEAR; ++j)
        s[j] = qa0 * kvflat[lane * K_NEAR + j] + qa1 * kvflat[(lane + 32) * K_NEAR + j];
#pragma unroll
    for (int o = 16; o > 0; o >>= 1) {
#pragma unroll
        for (int j = 0; j < K_NEAR; ++j)
            s[j] += __shfl_xor(s[j], o, 32);             // 10 independent chains per step
    }
    float m = s[0] * 0.125f;
#pragma unroll
    for (int j = 0; j < K_NEAR; ++j) { s[j] *= 0.125f; m = fmaxf(m, s[j]); }
    float e[K_NEAR], sum = 0.f;
#pragma unroll
    for (int j = 0; j < K_NEAR; ++j) { e[j] = expf(s[j] - m); sum += e[j]; }
    float inv = 1.0f / sum;
    float a0 = 0.f, a1 = 0.f;
#pragma unroll
    for (int j = 0; j < K_NEAR; ++j) {
        float w = e[j] * inv;
        a0 += w * kvflat[j * D4 + lane];                 // weighted sum uses true (K,d) layout
        a1 += w * kvflat[j * D4 + lane + 32];
    }
    out[lane]      = a0;
    out[lane + 32] = a1;
}

__global__ __launch_bounds__(256) void region_attn(const int* __restrict__ near,
                                                   const float* __restrict__ E_in,
                                                   const float* __restrict__ E_out,
                                                   float* __restrict__ region_out,
                                                   float* __restrict__ region_in) {
    __shared__ float sIn [8][K_NEAR * D4];
    __shared__ float sOut[8][K_NEAR * D4];
    int wave = threadIdx.x >> 5;
    int lane = threadIdx.x & 31;
    int n = blockIdx.x * 8 + wave;
    if (n >= ITEM_NUM) return;                           // wave-uniform
    float* fin  = sIn[wave];
    float* fout = sOut[wave];
#pragma unroll
    for (int j = 0; j < K_NEAR; ++j) {
        int p = near[n * K_NEAR + j];
        fin [j * D4 + lane]      = E_in [p * D4 + lane];
        fin [j * D4 + lane + 32] = E_in [p * D4 + lane + 32];
        fout[j * D4 + lane]      = E_out[p * D4 + lane];
        fout[j * D4 + lane + 32] = E_out[p * D4 + lane + 32];
    }
    // wave-local LDS: intra-wave DS ordering + compiler dscnt waits suffice
    attn_one(fin,  fout, region_out + n * D4, lane);     // q=ingoing, kv=outgoing
    attn_one(fout, fin,  region_in  + n * D4, lane);     // q=outgoing, kv=ingoing
}

// -------------------------------------------------------------------------
// Kernel 2: per-batch fused MLP + weighting. One block (8 waves) per b.
// X = (hist ⊙ targ) as fp16 in LDS (64x256 padded, rows 50..63 zero).
// Wave w -> N-tile w; 4 M-tiles x 8 K-steps of wmma_f32_16x16x32_f16.
// Epilogue: relu(acc+b1)*w2 -> LDS h1 buffer (plain stores), then one
// parallel row-reduction pass (no serialized bpermute chains).
// -------------------------------------------------------------------------
__global__ __launch_bounds__(256) void score_kernel(const int* __restrict__ history,
                                                    const int* __restrict__ target,
                                                    const float* __restrict__ E_hist,
                                                    const float* __restrict__ E_targ,
                                                    const float* __restrict__ region_in,
                                                    const float* __restrict__ region_out,
                                                    const _Float16* __restrict__ w1frag,
                                                    const float* __restrict__ b1,
                                                    const float* __restrict__ W2,
                                                    float* __restrict__ out) {
    constexpr int XS  = 264;                             // x row stride (halves), 16B-aligned
    constexpr int H1S = 132;                             // h1 row stride (floats), 16B-aligned
    __shared__ float sTarg[EMBED];
    __shared__ __align__(16) _Float16 sX[64 * XS];
    __shared__ __align__(16) float sH1[64 * H1S];
    __shared__ float sRowSum[64];
    __shared__ float sScore[64];

    int b = blockIdx.x;
    int tid  = threadIdx.x;
    int lane = tid & 31;
    int wave = tid >> 5;
    int t = target[b];

    {   // targ = [E_targ | region_out | region_in]
        int e = tid;
        float v;
        if (e < 128)      v = E_targ[t * D2 + e];
        else if (e < 192) v = region_out[t * D4 + (e - 128)];
        else              v = region_in [t * D4 + (e - 192)];
        sTarg[e] = v;
    }
    __syncthreads();

    // build X rows; hist = [E_hist | region_in | region_out]
    {
        float rsum[8];
#pragma unroll
        for (int hh = 0; hh < 8; ++hh) {
            int h = wave + hh * 8;                       // 0..63
            float rs = 0.f;
            if (h < HIST) {
                int idx = history[b * HIST + h];
#pragma unroll
                for (int rep = 0; rep < 8; ++rep) {
                    int e = lane + rep * 32;
                    float hv;
                    if (e < 128)      hv = E_hist[idx * D2 + e];
                    else if (e < 192) hv = region_in [idx * D4 + (e - 128)];
                    else              hv = region_out[idx * D4 + (e - 192)];
                    float p = hv * sTarg[e];
                    sX[h * XS + e] = (_Float16)p;
                    rs += p;
                }
            } else {
#pragma unroll
                for (int rep = 0; rep < 8; ++rep) sX[h * XS + lane + rep * 32] = (_Float16)0.f;
            }
            rsum[hh] = rs;
        }
#pragma unroll
        for (int o = 16; o > 0; o >>= 1) {
#pragma unroll
            for (int hh = 0; hh < 8; ++hh)
                rsum[hh] += __shfl_xor(rsum[hh], o, 32); // 8 independent chains per step
        }
        if (lane == 0) {
#pragma unroll
            for (int hh = 0; hh < 8; ++hh) sRowSum[wave + hh * 8] = rsum[hh];
        }
    }
    __syncthreads();

    {   // GEMM + fused bias/relu/w2; store scaled h1 (EXEC all-ones here)
        int ntile = wave;
        int c = lane & 15;                               // A row / B,D column
        int g = lane >> 4;
        v16h bfrag[8];
#pragma unroll
        for (int ks = 0; ks < 8; ++ks)
            bfrag[ks] = ((const v16h*)w1frag)[(ntile * 8 + ks) * 32 + lane];
        float w2v = W2[ntile * 16 + c];
        float b1v = b1[ntile * 16 + c];
#pragma unroll
        for (int mtile = 0; mtile < 4; ++mtile) {
            v8f acc = {};
            const _Float16* xrow = &sX[(mtile * 16 + c) * XS];
#pragma unroll
            for (int ks = 0; ks < 8; ++ks) {
                // A halves 0..7 -> K = ks*32 + g*8 + 0..7 ; 8..15 -> +16
                v8h lo = *(const v8h*)(xrow + ks * 32 + g * 8);
                v8h hi = *(const v8h*)(xrow + ks * 32 + 16 + g * 8);
                v16h a = __builtin_shufflevector(lo, hi, 0,1,2,3,4,5,6,7,8,9,10,11,12,13,14,15);
                acc = __builtin_amdgcn_wmma_f32_16x16x32_f16(false, a, false, bfrag[ks],
                                                             (short)0, acc, false, false);
            }
#pragma unroll
            for (int gg = 0; gg < 8; ++gg) {             // D: VGPR gg -> row gg (+8 for g=1)
                int row = mtile * 16 + g * 8 + gg;
                sH1[row * H1S + ntile * 16 + c] = fmaxf(acc[gg] + b1v, 0.f) * w2v;
            }
        }
    }
    __syncthreads();

    {   // parallel score reduction: 4 threads per row, 32 cols each
        int row = tid >> 2;                              // 0..63
        int q   = tid & 3;
        const float4* p4 = (const float4*)&sH1[row * H1S + q * 32];
        float s = 0.f;
#pragma unroll
        for (int k = 0; k < 8; ++k) {
            float4 v = p4[k];
            s += (v.x + v.y) + (v.z + v.w);
        }
        s += __shfl_xor(s, 1, 32);
        s += __shfl_xor(s, 2, 32);
        if (q == 0) sScore[row] = s;
    }
    __syncthreads();

    if (wave == 0) {                                     // masked exp / denom**0.5 / sigmoid
        float S1 = 0.f, S2 = 0.f;
#pragma unroll
        for (int rep = 0; rep < 2; ++rep) {
            int h = lane + rep * 32;
            if (h < HIST) {
                int idx = history[b * HIST + h];
                float eh = (idx != t) ? expf(sScore[h]) : 0.f;
                S1 += eh;
                S2 += eh * sRowSum[h];
            }
        }
#pragma unroll
        for (int o = 16; o > 0; o >>= 1) {
            S1 += __shfl_xor(S1, o, 32);
            S2 += __shfl_xor(S2, o, 32);
        }
        if (lane == 0) {
            float pred = S2 / sqrtf(S1);                 // denom = S1 ** 0.5 (BETA)
            out[b] = 1.f / (1.f + expf(-pred));
        }
    }
}

// -------------------------------------------------------------------------
extern "C" void kernel_launch(void* const* d_in, const int* in_sizes, int n_in,
                              void* d_out, int out_size, void* d_ws, size_t ws_size,
                              hipStream_t stream) {
    const int*   history = (const int*)  d_in[0];
    const int*   target  = (const int*)  d_in[1];
    const int*   near    = (const int*)  d_in[2];
    // d_in[3] target_region: unused by the reference
    const float* E_in    = (const float*)d_in[4];
    const float* E_out   = (const float*)d_in[5];
    const float* E_hist  = (const float*)d_in[6];
    const float* E_targ  = (const float*)d_in[7];
    const float* W1      = (const float*)d_in[8];
    const float* b1      = (const float*)d_in[9];
    const float* W2      = (const float*)d_in[10];
    float* out = (float*)d_out;

    char* ws = (char*)d_ws;
    float*    region_out = (float*)ws;                      // 50000*64*4 = 12.8 MB
    float*    region_in  = (float*)(ws + 12800000);         // 12.8 MB
    _Float16* w1frag     = (_Float16*)(ws + 25600000);      // 64 KB

    w1_to_frag <<<128, 256, 0, stream>>>(W1, w1frag);
    region_attn<<<(ITEM_NUM + 7) / 8, 256, 0, stream>>>(near, E_in, E_out,
                                                        region_out, region_in);
    score_kernel<<<BATCH, 256, 0, stream>>>(history, target, E_hist, E_targ,
                                            region_in, region_out, w1frag, b1, W2, out);
}
